// TGAGraphAttention_90168543412704
// MI455X (gfx1250) — matrix operand order repaired and checked by reading, not compiled, not run on hardware
//
#include <hip/hip_runtime.h>
#include <hip/hip_bf16.h>

typedef __attribute__((ext_vector_type(16))) _Float16 v16h;
typedef __attribute__((ext_vector_type(8)))  float    v8f;

#define NHEAD 4
#define DIN   32
#define DOUT  32
#define NEG_SLOPE 0.2f

// ---- monotone float<->uint encoding for atomicMax on floats (handles negatives) ----
__device__ __forceinline__ unsigned fenc(float f) {
    unsigned u = __float_as_uint(f);
    return (u & 0x80000000u) ? ~u : (u | 0x80000000u);
}
__device__ __forceinline__ float fdec(unsigned u) {
    return (u & 0x80000000u) ? __uint_as_float(u & 0x7fffffffu)
                             : __uint_as_float(~u);
}
// fenc(-inf) == 0x007FFFFF

// wave-uniform lane broadcast -> v_readlane_b32 (SGPR result)
__device__ __forceinline__ int bcast_i(int v, int lane) {
    return __builtin_amdgcn_readlane(v, lane);
}
__device__ __forceinline__ float bcast_f(float v, int lane) {
    return __uint_as_float((unsigned)__builtin_amdgcn_readlane((int)__float_as_uint(v), lane));
}

// ------------------------------------------------------------------
// Init: zero accumulators/denoms, set running-max to encoded -inf
// ------------------------------------------------------------------
__global__ void gat_init_kernel(float* __restrict__ acc, float* __restrict__ denom,
                                unsigned* __restrict__ emax, long accN, int nodeTot) {
    long i = blockIdx.x * (long)blockDim.x + threadIdx.x;
    long stride = (long)gridDim.x * blockDim.x;
    for (long j = i; j < accN; j += stride) acc[j] = 0.0f;
    for (long j = i; j < nodeTot; j += stride) { denom[j] = 0.0f; emax[j] = 0x007FFFFFu; }
}

// ------------------------------------------------------------------
// Phase 1: xl = x @ W_l, xr = x @ W_r per head via WMMA f16->f32.
// One wave per (head, 16-node tile); K=32 == one v_wmma_f32_16x16x32_f16
// per 16x16 output tile (4 WMMAs per wave: {W_l,W_r} x {cols 0-15,16-31}).
// ------------------------------------------------------------------
__global__ __launch_bounds__(256)
void gat_transform_kernel(const float* __restrict__ x,
                          const float* __restrict__ Wl,
                          const float* __restrict__ Wr,
                          float* __restrict__ xl, float* __restrict__ xr,
                          int N, int ntiles) {
    int wid  = (int)((blockIdx.x * (long)blockDim.x + threadIdx.x) >> 5);
    int lane = threadIdx.x & 31;
    if (wid >= NHEAD * ntiles) return;
    int h = wid / ntiles;
    int t = wid - h * ntiles;

    int half = lane >> 4;        // which half-wave
    int lrow = lane & 15;
    bool fullTile = (t * 16 + 16 <= N);   // wave-uniform

    // A-matrix (16x32 f16): lanes 0-15 hold K {0..7,16..23}; lanes 16-31 hold {8..15,24..31}
    int klo = half ? 8  : 0;
    int khi = half ? 24 : 16;
    int node = t * 16 + lrow;
    if (node >= N) node = N - 1;
    const float* xrow = x + ((size_t)h * N + node) * DIN;
    v16h a;
#pragma unroll
    for (int i = 0; i < 8; ++i) {
        a[i]     = (_Float16)xrow[klo + i];
        a[8 + i] = (_Float16)xrow[khi + i];
    }

    const float* Wmat[2] = { Wl + (size_t)h * DIN * DOUT, Wr + (size_t)h * DIN * DOUT };
    float*       Obuf[2] = { xl + (size_t)h * N * DOUT,   xr + (size_t)h * N * DOUT };

#pragma unroll
    for (int m = 0; m < 2; ++m) {
#pragma unroll
        for (int cb = 0; cb < 2; ++cb) {
            int c0  = cb * 16;
            int col = c0 + lrow;
            // B-matrix (32x16 f16): lanes 0-15 hold K {0..7,16..23}; lanes 16-31 hold {8..15,24..31}
            int kb = half ? 16 : 0;
            v16h b;
#pragma unroll
            for (int i = 0; i < 16; ++i)
                b[i] = (_Float16)Wmat[m][(size_t)((i < 8) ? (kb / 2 + i) : (8 + kb / 2 + i)) * DOUT + col];

            v8f c = {};
            c = __builtin_amdgcn_wmma_f32_16x16x32_f16(
                    false, a, false, b, (short)0, c, false, false);

            // D layout: VGPR r -> row M = 8*half + r, col N = lrow
            float* obase = Obuf[m] + (size_t)(t * 16 + half * 8) * DOUT + c0 + lrow;
            if (fullTile) {
#pragma unroll
                for (int r = 0; r < 8; ++r)
                    obase[(size_t)r * DOUT] = c[r];   // constant 128B offsets off one base
            } else {
#pragma unroll
                for (int r = 0; r < 8; ++r) {
                    int nn = t * 16 + half * 8 + r;
                    if (nn < N) Obuf[m][(size_t)nn * DOUT + c0 + lrow] = c[r];
                }
            }
        }
    }
}

// ------------------------------------------------------------------
// Phase 2: per-edge attention logit + running max per destination.
// Lane-per-edge; xl/xr rows are L2-resident 128B gathers (b128 loads).
// ------------------------------------------------------------------
__global__ void gat_logits_kernel(const long long* __restrict__ ei,
                                  const float* __restrict__ xl,
                                  const float* __restrict__ xr,
                                  const float* __restrict__ att,
                                  float* __restrict__ elog,
                                  unsigned* __restrict__ emax,
                                  int N, int E, int Ep) {
    long i = blockIdx.x * (long)blockDim.x + threadIdx.x;
    if (i >= (long)NHEAD * Ep) return;
    int h = (int)(i / Ep);
    int e = (int)(i - (long)h * Ep);

    int s, d;
    if (e < E) {
        const long long* base = ei + (size_t)h * 2 * E;   // [2][E] per head (int64)
        s = (int)base[e];
        d = (int)base[E + e];
    } else {
        s = d = e - E;                                    // self-loop
    }

    const float4* pl = (const float4*)(xl + ((size_t)h * N + s) * DOUT);
    const float4* pr = (const float4*)(xr + ((size_t)h * N + d) * DOUT);
    const float4* pa = (const float4*)(att + (size_t)h * DOUT);
    float sum = 0.0f;
#pragma unroll
    for (int q = 0; q < 8; ++q) {
        float4 a4 = pl[q], b4 = pr[q], t4 = pa[q];
        float v;
        v = a4.x + b4.x; v = v > 0.0f ? v : v * NEG_SLOPE; sum += v * t4.x;
        v = a4.y + b4.y; v = v > 0.0f ? v : v * NEG_SLOPE; sum += v * t4.y;
        v = a4.z + b4.z; v = v > 0.0f ? v : v * NEG_SLOPE; sum += v * t4.z;
        v = a4.w + b4.w; v = v > 0.0f ? v : v * NEG_SLOPE; sum += v * t4.w;
    }
    elog[i] = sum;
    atomicMax(&emax[(size_t)h * N + d], fenc(sum));
}

// ------------------------------------------------------------------
// Phase 3: w = exp(e - max[dst]); denom[dst] += w; acc[dst] += w*xl[src].
// Wave handles 32 edges: per-lane w/denom, then lanes-as-features so each
// edge's 32-feature scatter is ONE coalesced 128B global_atomic_add_f32.
// src/dst/w broadcast with v_readlane -> scalar bases (SADDR addressing).
// ------------------------------------------------------------------
__global__ __launch_bounds__(256)
void gat_scatter_kernel(const long long* __restrict__ ei,
                        const float* __restrict__ xl,
                        const float* __restrict__ elog,
                        const unsigned* __restrict__ emax,
                        float* __restrict__ denom,
                        float* __restrict__ acc,
                        int N, int E, int Ep, int chunksPerHead) {
    int wid  = (int)((blockIdx.x * (long)blockDim.x + threadIdx.x) >> 5);
    int lane = threadIdx.x & 31;
    if (wid >= NHEAD * chunksPerHead) return;
    int h = wid / chunksPerHead;
    int c = wid - h * chunksPerHead;
    int base = c * 32;

    int s = 0, d = 0;
    float w = 0.0f;
    int e = base + lane;
    if (e < Ep) {
        if (e < E) {
            const long long* bp = ei + (size_t)h * 2 * E;
            s = (int)bp[e];
            d = (int)bp[E + e];
        } else {
            s = d = e - E;
        }
        float m = fdec(emax[(size_t)h * N + d]);
        w = __expf(elog[(size_t)h * Ep + e] - m);
        atomicAdd(&denom[(size_t)h * N + d], w);
    }

    int nvalid = Ep - base; if (nvalid > 32) nvalid = 32;
    const float* xlh  = xl  + (size_t)h * N * DOUT;
    float*       acch = acc + (size_t)h * N * DOUT;
    for (int j = 0; j < nvalid; ++j) {
        int   sj = bcast_i(s, j);                 // v_readlane -> SGPR
        int   dj = bcast_i(d, j);
        float wj = bcast_f(w, j);
        float v = wj * xlh[(size_t)sj * DOUT + lane];   // coalesced 128B row (scalar base)
        atomicAdd(&acch[(size_t)dj * DOUT + lane], v);  // coalesced 128B atomic
    }
}

// ------------------------------------------------------------------
// Phase 4: out[n, h*32+k] = acc[h][n][k] / denom[h][n] + bias[h][k]
// ------------------------------------------------------------------
__global__ void gat_finalize_kernel(const float* __restrict__ acc,
                                    const float* __restrict__ denom,
                                    const float* __restrict__ bias,
                                    float* __restrict__ out, int N) {
    long i = blockIdx.x * (long)blockDim.x + threadIdx.x;
    if (i >= (long)N * (NHEAD * DOUT)) return;
    int n = (int)(i / (NHEAD * DOUT));
    int r = (int)(i - (long)n * (NHEAD * DOUT));
    int h = r / DOUT;
    int k = r - h * DOUT;
    float dn = denom[(size_t)h * N + n];
    out[i] = acc[((size_t)h * N + n) * DOUT + k] / dn + bias[h * DOUT + k];
}

// ------------------------------------------------------------------
extern "C" void kernel_launch(void* const* d_in, const int* in_sizes, int n_in,
                              void* d_out, int out_size, void* d_ws, size_t ws_size,
                              hipStream_t stream) {
    const float*     x    = (const float*)d_in[0];
    const long long* ei   = (const long long*)d_in[1];   // int64 edge_index [H,2,E]
    const float*     Wl   = (const float*)d_in[2];
    const float*     Wr   = (const float*)d_in[3];
    const float*     att  = (const float*)d_in[4];
    const float*     bias = (const float*)d_in[5];
    float*           out  = (float*)d_out;

    int N  = in_sizes[0] / (NHEAD * DIN);
    int E  = in_sizes[1] / (NHEAD * 2);
    int Ep = E + N;                       // edges + self-loops

    // Workspace layout (~105 MB total for N=50k, E=1.6M)
    char* ws = (char*)d_ws;
    size_t off = 0;
    float*    xl    = (float*)(ws + off); off += (size_t)NHEAD * N * DOUT * 4;
    float*    xr    = (float*)(ws + off); off += (size_t)NHEAD * N * DOUT * 4;
    float*    elog  = (float*)(ws + off); off += (size_t)NHEAD * Ep * 4;
    unsigned* emax  = (unsigned*)(ws + off); off += (size_t)NHEAD * N * 4;
    float*    denom = (float*)(ws + off); off += (size_t)NHEAD * N * 4;
    float*    acc   = (float*)(ws + off); off += (size_t)NHEAD * N * DOUT * 4;
    (void)ws_size; (void)n_in; (void)out_size;

    // 0) init accumulators
    gat_init_kernel<<<2048, 256, 0, stream>>>(acc, denom, emax,
                                              (long)NHEAD * N * DOUT, NHEAD * N);
    // 1) WMMA feature transforms
    {
        int ntiles = (N + 15) / 16;
        int waves  = NHEAD * ntiles;
        gat_transform_kernel<<<(waves + 7) / 8, 256, 0, stream>>>(x, Wl, Wr, xl, xr, N, ntiles);
    }
    // 2) attention logits + per-destination running max
    {
        long tot = (long)NHEAD * Ep;
        gat_logits_kernel<<<(int)((tot + 255) / 256), 256, 0, stream>>>(
            ei, xl, xr, att, elog, emax, N, E, Ep);
    }
    // 3) exp + denom + weighted feature scatter (L2 atomics)
    {
        int cph   = (Ep + 31) / 32;
        int waves = NHEAD * cph;
        gat_scatter_kernel<<<(waves + 7) / 8, 256, 0, stream>>>(
            ei, xl, elog, emax, denom, acc, N, E, Ep, cph);
    }
    // 4) normalize + bias, interleave heads into [N, H*DOUT]
    {
        long tot = (long)N * NHEAD * DOUT;
        gat_finalize_kernel<<<(int)((tot + 255) / 256), 256, 0, stream>>>(
            acc, denom, bias, out, N);
    }
}
